// SelfAttention_88235808129071
// MI455X (gfx1250) — matrix operand order repaired and checked by reading, not compiled, not run on hardware
//
#include <hip/hip_runtime.h>
#include <hip/hip_bf16.h>
#include <math.h>

typedef __attribute__((ext_vector_type(16))) _Float16 v16h;
typedef __attribute__((ext_vector_type(8)))  _Float16 v8h;
typedef __attribute__((ext_vector_type(8)))  float    v8f;
typedef __attribute__((ext_vector_type(4)))  float    v4f;

#define EMBED 1024
#define HEADS 16
#define HD    64
#define BATCH 4
#define SEQ   1024
#define SCALE (1.0f / 32.0f)   // 1/sqrt(EMBED)
#define USE_ASYNC 1

__device__ __forceinline__ v8f wmma16(v16h a, v16h b, v8f c) {
  return __builtin_amdgcn_wmma_f32_16x16x32_f16(false, a, false, b, (short)0, c,
                                                false, false);
}

// LDS byte offset of a generic pointer into __shared__ (AS3 offset = addr[31:0])
__device__ __forceinline__ unsigned lds_off(const void* p) {
  return (unsigned)(unsigned long long)p;
}

#if USE_ASYNC
// GVS-mode async global->LDS 16B copy: per-lane LDS addr + per-lane 32b offset
// against a uniform 64-bit SGPR base. Tracked by ASYNCcnt.
__device__ __forceinline__ void async_b128(unsigned lds, unsigned voff,
                                           const void* base) {
  asm volatile("global_load_async_to_lds_b128 %0, %1, %2"
               :: "v"(lds), "v"(voff), "s"((unsigned long long)base)
               : "memory");
}
__device__ __forceinline__ void wait_async0() {
  asm volatile("s_wait_asynccnt 0x0" ::: "memory");
}
#endif

// A-matrix tile (16x32 f16): lane = M row; K set {off0..off0+7, off0+16..off0+23}
__device__ __forceinline__ v16h a_from_f16(const _Float16* p, int off0) {
  v8h x0 = *(const v8h*)(p + off0);
  v8h x1 = *(const v8h*)(p + off0 + 16);
  v16h a;
#pragma unroll
  for (int i = 0; i < 8; ++i) { a[i] = x0[i]; a[8 + i] = x1[i]; }
  return a;
}
__device__ __forceinline__ v16h a_from_f32(const float* p, int off0) {
  v4f x0 = *(const v4f*)(p + off0);
  v4f x1 = *(const v4f*)(p + off0 + 4);
  v4f x2 = *(const v4f*)(p + off0 + 16);
  v4f x3 = *(const v4f*)(p + off0 + 20);
  v16h a;
#pragma unroll
  for (int i = 0; i < 4; ++i) {
    a[i]      = (_Float16)x0[i];
    a[4 + i]  = (_Float16)x1[i];
    a[8 + i]  = (_Float16)x2[i];
    a[12 + i] = (_Float16)x3[i];
  }
  return a;
}
// B-matrix tile: lane = N column; K rows {0..7, 16..23} from p (row stride ld)
__device__ __forceinline__ v16h ld16(const _Float16* p, int ld) {
  v16h a;
#pragma unroll
  for (int i = 0; i < 8; ++i) { a[i] = p[i * ld]; a[8 + i] = p[(16 + i) * ld]; }
  return a;
}

// ---------------------------------------------------------------------------
// Kernel A: per-head QKV projection via WMMA. W^T staged (transposed) in LDS.
// Writes qh[n,h,l,d], kT[n,h,d,l], vh[n,h,l,d] f16.
// ---------------------------------------------------------------------------
__global__ void __launch_bounds__(128) qkv_proj_kernel(
    const float* __restrict__ Vx, const float* __restrict__ Qx,
    const float* __restrict__ Kx,
    const float* __restrict__ Wv, const float* __restrict__ Wq,
    const float* __restrict__ Wk,
    _Float16* __restrict__ qh, _Float16* __restrict__ kT,
    _Float16* __restrict__ vh) {
  __shared__ __attribute__((aligned(16))) _Float16 sWT[3][64][64]; // WT[d][e]
  const int tid  = threadIdx.x;
  const int lane = tid & 31;
  const int wid  = blockIdx.x * 4 + (tid >> 5);
  const int h    = wid & (HEADS - 1);
  const int tile = wid >> 4;
  const int row0 = tile * 16;
  const int m    = lane & 15;
  const bool hi  = lane >= 16;
  const int off0 = hi ? 8 : 0;
  const int klane = m + (hi ? 16 : 0);
  const int n  = row0 >> 10;
  const int l0 = row0 & (SEQ - 1);
  const size_t hb = (size_t)(n * HEADS + h) * SEQ;

  {
    const float* Ws[3] = {Wq, Wk, Wv};
#pragma unroll
    for (int w = 0; w < 3; ++w)
      for (int idx = tid; idx < 64 * 64; idx += 128) {
        const int e = idx >> 6, d = idx & 63;
        sWT[w][d][e] = (_Float16)Ws[w][idx];   // idx = e*64 + d
      }
  }
  __syncthreads();

  for (int which = 0; which < 3; ++which) {
    const float* X = (which == 0) ? Qx : (which == 1) ? Kx : Vx;
    const float* xp = X + (size_t)(row0 + m) * EMBED + h * HD;
    v16h a0 = a_from_f32(xp, off0);
    v16h a1 = a_from_f32(xp + 32, off0);
#pragma unroll
    for (int t = 0; t < 4; ++t) {
      v16h b0 = ld16(&sWT[which][off0][t * 16 + m], 64);
      v16h b1 = ld16(&sWT[which][32 + off0][t * 16 + m], 64);
      v8f c;
#pragma unroll
      for (int j = 0; j < 8; ++j) c[j] = 0.f;
      c = wmma16(a0, b0, c);
      c = wmma16(a1, b1, c);
      const int ecol = t * 16 + m;
      if (which == 1) {                       // kT: 8 contiguous f16 -> b128
        v8h kv;
#pragma unroll
        for (int j = 0; j < 8; ++j) kv[j] = (_Float16)c[j];
        _Float16* p = kT + ((size_t)(n * HEADS + h) * HD + ecol) * SEQ + l0 +
                      (hi ? 8 : 0);
        *(v8h*)p = kv;
      } else {
        _Float16* dst = (which == 0) ? qh : vh;
#pragma unroll
        for (int j = 0; j < 8; ++j) {
          const int l = l0 + j + (hi ? 8 : 0);
          dst[(hb + l) * HD + ecol] = (_Float16)c[j];
        }
      }
    }
  }
}

// ---------------------------------------------------------------------------
// Kernel B: flash attention. Block = one (n,h), 4 waves x 16-query tiles.
// K/V chunks double-buffered in LDS; async fill of chunk i+1 overlaps the
// WMMA work on chunk i (single barrier per chunk, waits only on ASYNCcnt).
// ---------------------------------------------------------------------------
__global__ void __launch_bounds__(128) attn_kernel(
    const _Float16* __restrict__ qh, const _Float16* __restrict__ kT,
    const _Float16* __restrict__ vh, const int* __restrict__ mask,
    _Float16* __restrict__ xout) {
  __shared__ __attribute__((aligned(16))) _Float16 sK[2][64][32];  // [d][key]
  __shared__ __attribute__((aligned(16))) _Float16 sV[2][32][64];  // [key][d]
  __shared__ __attribute__((aligned(16))) _Float16 sP[4][16][32];  // per-wave
  const int tid   = threadIdx.x;
  const int lane  = tid & 31;
  const int wslot = tid >> 5;
  const int n  = blockIdx.x >> 8;
  const int h  = (blockIdx.x >> 4) & 15;
  const int qg = blockIdx.x & 15;
  const int q0 = qg * 64 + wslot * 16;
  const int m  = lane & 15;
  const bool hi = lane >= 16;
  const int off0  = hi ? 8 : 0;
  const int klane = m + (hi ? 16 : 0);
  const int hbase = n * HEADS + h;

  const _Float16* kTp = kT + (size_t)hbase * HD * SEQ;
  const _Float16* vp  = vh + (size_t)hbase * SEQ * HD;
  const int*      mp  = mask + (size_t)n * SEQ * SEQ;

  const _Float16* qp = qh + ((size_t)hbase * SEQ + q0 + m) * HD;
  v16h qa0 = a_from_f16(qp, off0);
  v16h qa1 = a_from_f16(qp + 32, off0);

  // stage K chunk [64 d][32 key] and V chunk [32 key][64 d] into buffer `buf`
  auto stage = [&](int kt, int buf) {
#if USE_ASYNC
    const unsigned lK = lds_off(&sK[buf][0][0]);
    const unsigned lV = lds_off(&sV[buf][0][0]);
#pragma unroll
    for (int i = 0; i < 2; ++i) {              // 256 x 16B chunks for sK
      const int c = tid * 2 + i;
      const int d = c >> 2, part = c & 3;
      async_b128(lK + (unsigned)(d * 64 + part * 16),
                 (unsigned)(((size_t)d * SEQ + kt + part * 8) * 2), kTp);
    }
#pragma unroll
    for (int i = 0; i < 2; ++i) {              // 256 x 16B chunks for sV
      const int c = tid * 2 + i;
      const int kk = c >> 3, part = c & 7;
      async_b128(lV + (unsigned)(kk * 128 + part * 16),
                 (unsigned)(((size_t)(kt + kk) * HD + part * 8) * 2), vp);
    }
#else
#pragma unroll
    for (int i = 0; i < 2; ++i) {
      const int c = tid * 2 + i;
      const int d = c >> 2, part = c & 3;
      *(v8h*)&sK[buf][d][part * 8] =
          *(const v8h*)(kTp + (size_t)d * SEQ + kt + part * 8);
      const int kk = c >> 3, p2 = c & 7;
      *(v8h*)&sV[buf][kk][p2 * 8] =
          *(const v8h*)(vp + (size_t)(kt + kk) * HD + p2 * 8);
    }
#endif
  };

  float mrow[8], lrow[8];
  v8f acc[4];
#pragma unroll
  for (int j = 0; j < 8; ++j) { mrow[j] = -__builtin_inff(); lrow[j] = 0.f; }
#pragma unroll
  for (int t = 0; t < 4; ++t)
#pragma unroll
    for (int j = 0; j < 8; ++j) acc[t][j] = 0.f;

  stage(0, 0);                                 // prime the pipeline
  for (int it = 0; it < SEQ / 32; ++it) {
    const int kt = it * 32;
    const int b  = it & 1;
#if USE_ASYNC
    wait_async0();                             // my chunk-it fills are done
#endif
    __syncthreads();                           // all waves: chunk ready, buf
                                               // b^1 fully consumed last iter
    if (kt + 32 < SEQ) stage(kt + 32, b ^ 1);  // overlaps with compute below

    // ---- scores: S = Q K^T (two 16-key subtiles) ----
    float tt[2][8];
#pragma unroll
    for (int sub = 0; sub < 2; ++sub) {
      v16h b0 = ld16(&sK[b][off0][sub * 16 + m], 32);
      v16h b1 = ld16(&sK[b][32 + off0][sub * 16 + m], 32);
      v8f s;
#pragma unroll
      for (int j = 0; j < 8; ++j) s[j] = 0.f;
      s = wmma16(qa0, b0, s);
      s = wmma16(qa1, b1, s);
      const int col = kt + sub * 16 + m;
#pragma unroll
      for (int j = 0; j < 8; ++j) {
        const int qrow = q0 + j + (hi ? 8 : 0);
        const int mk = mp[(size_t)qrow * SEQ + col];
        tt[sub][j] = (mk == 0) ? -__builtin_inff() : s[j] * SCALE;
      }
    }
    // ---- online softmax (rows across 16 lanes of each half-wave) ----
    _Float16 pf[2][8];
#pragma unroll
    for (int j = 0; j < 8; ++j) {
      float rmax = fmaxf(tt[0][j], tt[1][j]);
#pragma unroll
      for (int o = 8; o >= 1; o >>= 1)
        rmax = fmaxf(rmax, __shfl_xor(rmax, o, 32));
      const float mnew = fmaxf(mrow[j], rmax);
      const float corr =
          (mrow[j] == -__builtin_inff()) ? 0.f : __expf(mrow[j] - mnew);
      const float p0 = (tt[0][j] == -__builtin_inff()) ? 0.f : __expf(tt[0][j] - mnew);
      const float p1 = (tt[1][j] == -__builtin_inff()) ? 0.f : __expf(tt[1][j] - mnew);
      float rs = p0 + p1;
#pragma unroll
      for (int o = 8; o >= 1; o >>= 1)
        rs += __shfl_xor(rs, o, 32);
      lrow[j] = lrow[j] * corr + rs;
      mrow[j] = mnew;
#pragma unroll
      for (int t = 0; t < 4; ++t) acc[t][j] *= corr;
      pf[0][j] = (_Float16)p0;
      pf[1][j] = (_Float16)p1;
    }
    // ---- relayout P: C layout -> A layout via per-wave LDS slot ----
    __builtin_amdgcn_wave_barrier();
#pragma unroll
    for (int sub = 0; sub < 2; ++sub)
#pragma unroll
      for (int j = 0; j < 8; ++j)
        sP[wslot][j + (hi ? 8 : 0)][sub * 16 + m] = pf[sub][j];
    __builtin_amdgcn_wave_barrier();
    v16h pa = a_from_f16(&sP[wslot][m][0], off0);
    __builtin_amdgcn_wave_barrier();
    // ---- acc += P x V ----
#pragma unroll
    for (int t = 0; t < 4; ++t) {
      v16h bv = ld16(&sV[b][off0][t * 16 + m], 64);
      acc[t] = wmma16(pa, bv, acc[t]);
    }
  }
  // ---- finalize ----
#pragma unroll
  for (int j = 0; j < 8; ++j) {
    const float inv = 1.0f / lrow[j];
    const int qrow = q0 + j + (hi ? 8 : 0);
#pragma unroll
    for (int t = 0; t < 4; ++t) {
      const int e = h * HD + t * 16 + m;
      xout[((size_t)n * SEQ + qrow) * EMBED + e] = (_Float16)(acc[t][j] * inv);
    }
  }
}

// ---------------------------------------------------------------------------
// Wo transpose + f16 convert: WoT[k][e] = Wo[e][k]
// ---------------------------------------------------------------------------
__global__ void __launch_bounds__(256) wo_transpose_kernel(
    const float* __restrict__ Wo, _Float16* __restrict__ WoT) {
  const int idx = blockIdx.x * blockDim.x + threadIdx.x;
  const int e = idx >> 10;
  const int k = idx & (EMBED - 1);
  WoT[(size_t)k * EMBED + e] = (_Float16)Wo[(size_t)e * EMBED + k];
}

// ---------------------------------------------------------------------------
// Kernel C: out = X @ Wo^T + bo. Block = 4 row-tiles sharing one 64-col tile;
// Wo chunks double-buffered in LDS with async fill overlapping WMMA.
// ---------------------------------------------------------------------------
__global__ void __launch_bounds__(128) out_proj_kernel(
    const _Float16* __restrict__ X, const _Float16* __restrict__ WoT,
    const float* __restrict__ bo, float* __restrict__ out) {
  __shared__ __attribute__((aligned(16))) _Float16 sB[2][32][64];  // 8KB
  const int tid   = threadIdx.x;
  const int lane  = tid & 31;
  const int wslot = tid >> 5;
  const int ct  = blockIdx.x & 15;
  const int rtg = blockIdx.x >> 4;            // 0..63
  const int row0 = (rtg * 4 + wslot) * 16;
  const int col0 = ct * 64;
  const int m   = lane & 15;
  const bool hi = lane >= 16;
  const int off0  = hi ? 8 : 0;
  const int klane = m + (hi ? 16 : 0);

  auto stageB = [&](int k, int buf) {
#if USE_ASYNC
    const unsigned lB = lds_off(&sB[buf][0][0]);
#pragma unroll
    for (int i = 0; i < 2; ++i) {             // 256 x 16B chunks
      const int cc = tid * 2 + i;
      const int kk = cc >> 3, part = cc & 7;
      async_b128(lB + (unsigned)(kk * 128 + part * 16),
                 (unsigned)(((size_t)(k + kk) * EMBED + col0 + part * 8) * 2),
                 WoT);
    }
#else
#pragma unroll
    for (int i = 0; i < 2; ++i) {
      const int cc = tid * 2 + i;
      const int kk = cc >> 3, part = cc & 7;
      *(v8h*)&sB[buf][kk][part * 8] =
          *(const v8h*)(WoT + (size_t)(k + kk) * EMBED + col0 + part * 8);
    }
#endif
  };

  v8f c[4];
#pragma unroll
  for (int t = 0; t < 4; ++t)
#pragma unroll
    for (int j = 0; j < 8; ++j) c[t][j] = 0.f;

  const _Float16* xp = X + (size_t)(row0 + m) * EMBED;
  stageB(0, 0);
  for (int it = 0; it < EMBED / 32; ++it) {
    const int k = it * 32;
    const int b = it & 1;
#if USE_ASYNC
    wait_async0();
#endif
    __syncthreads();
    if (k + 32 < EMBED) stageB(k + 32, b ^ 1);
    v16h a = a_from_f16(xp + k, off0);
#pragma unroll
    for (int t = 0; t < 4; ++t) {
      v16h bb = ld16(&sB[b][off0][t * 16 + m], 64);
      c[t] = wmma16(a, bb, c[t]);
    }
  }
#pragma unroll
  for (int t = 0; t < 4; ++t) {
    const int e = col0 + t * 16 + m;
    const float bias = bo[e];
#pragma unroll
    for (int j = 0; j < 8; ++j) {
      const int row = row0 + j + (hi ? 8 : 0);
      out[(size_t)row * EMBED + e] = c[t][j] + bias;
    }
  }
}

extern "C" void kernel_launch(void* const* d_in, const int* in_sizes, int n_in,
                              void* d_out, int out_size, void* d_ws, size_t ws_size,
                              hipStream_t stream) {
  const float* values = (const float*)d_in[0];
  const float* query  = (const float*)d_in[1];
  const float* key    = (const float*)d_in[2];
  const int*   mask   = (const int*)d_in[3];
  const float* Wv = (const float*)d_in[4];
  const float* Wk = (const float*)d_in[5];
  const float* Wq = (const float*)d_in[6];
  const float* Wo = (const float*)d_in[7];
  const float* bo = (const float*)d_in[8];

  _Float16* ws = (_Float16*)d_ws;
  const size_t NHL = (size_t)BATCH * HEADS * SEQ * HD;
  _Float16* qh   = ws;
  _Float16* kT   = qh + NHL;
  _Float16* vh   = kT + NHL;
  _Float16* xout = vh + NHL;
  _Float16* WoT  = xout + (size_t)BATCH * SEQ * EMBED;

  qkv_proj_kernel<<<1024, 128, 0, stream>>>(values, query, key, Wv, Wq, Wk,
                                            qh, kT, vh);
  wo_transpose_kernel<<<(EMBED * EMBED) / 256, 256, 0, stream>>>(Wo, WoT);
  attn_kernel<<<1024, 128, 0, stream>>>(qh, kT, vh, mask, xout);
  out_proj_kernel<<<1024, 128, 0, stream>>>(xout, WoT, bo, (float*)d_out);
}
